// EnhancedLayeredGraphRNN_5351529250824
// MI455X (gfx1250) — compile-verified, matrix operand order checked by reading, not run on hardware
//
#include <hip/hip_runtime.h>
#include <hip/hip_bf16.h>

// ---------------- problem constants (from reference) ----------------
#define B_    4
#define T_    12
#define N_    5000
#define FIN_  32
#define HID_  128
#define L_    2
#define H_    2
#define E_    40000
#define BN_   (B_ * N_)          // 20000
#define GHID_ (H_ * HID_)        // 256
#define G3_   (3 * HID_)         // 384
#define NEG_SLOPE 0.2f
#define LN_EPS    1e-5f
#define UNITS_ (B_ * E_ * H_)    // 320000 edge-head units

// GEMM block tiling: 160 threads = 5 waves; block tile 80(M) x 64(N), K-step 32.
// 20000 % 80 == 0 and all N in {128,256,384} % 64 == 0 -> no partial tiles,
// so EXEC is all-ones at every WMMA (wave-uniform control flow only).
#define TM_ 80
#define TN_ 64
#define TK_ 32
#define GEMM_THREADS 160

typedef __bf16 bf16_t;
typedef __attribute__((ext_vector_type(16))) __bf16 v16bf;
typedef __attribute__((ext_vector_type(8)))  __bf16 v8bf;
typedef __attribute__((ext_vector_type(8)))  float  v8f;

// ---------------- weight conversion into WMMA B-fragment layout ----------------
// Logical B is (K x N) with B(k,n) = transposed ? src[n*K + k] : src[k*N + n].
// Fragment layout (per CDNA5 ISA 7.12.2, 16-bit operands, wave32):
//   tile (kt = k/32, ntile = n/16); lane = (n%16) + 16*((k>>3)&1);
//   j = (k&7) + 8*(k>>4)  -> dst[((kt*(N/16) + ntile)*32 + lane)*16 + j]
// Each lane's 16 values are then one contiguous 32B run -> 2x global_load_b128.
__global__ void conv_bf16_frag(const float* __restrict__ src, bf16_t* __restrict__ dst,
                               int K, int N, int transposed) {
    int i = blockIdx.x * blockDim.x + threadIdx.x;
    if (i >= K * N) return;
    int k = i / N, n = i - k * N;
    float v = transposed ? src[(size_t)n * K + k] : src[(size_t)k * N + n];
    int kt = k >> 5, kk = k & 31;
    int ntile = n >> 4, nn = n & 15;
    int lane = nn + (((kk >> 3) & 1) << 4);
    int j = (kk & 7) + ((kk >> 4) << 3);
    dst[(((size_t)kt * (N >> 4) + ntile) * 32 + lane) * 16 + j] = (bf16_t)v;
}

__device__ __forceinline__ v8bf cvt8(float4 a, float4 b) {
    v8bf r;
    r[0] = (bf16_t)a.x; r[1] = (bf16_t)a.y; r[2] = (bf16_t)a.z; r[3] = (bf16_t)a.w;
    r[4] = (bf16_t)b.x; r[5] = (bf16_t)b.y; r[6] = (bf16_t)b.z; r[7] = (bf16_t)b.w;
    return r;
}

// ---------------- LDS-tiled WMMA GEMM: C(M,N) = A(M,K;lda) @ B(K,N) + bias(N) ----
// A fp32 -> bf16 staged through LDS (shared conversion across 5 waves);
// B read directly from global in pre-swizzled fragment layout (L2-resident).
// Each wave owns a 16-row strip: 1 A fragment reused by 4 WMMAs per K-step.
__global__ __launch_bounds__(GEMM_THREADS)
void wmma_gemm_bias(const float* __restrict__ A, int lda,
                    const bf16_t* __restrict__ Bf,
                    const float* __restrict__ bias,
                    float* __restrict__ C, int M, int N, int K) {
    __shared__ __align__(16) bf16_t Asb[TM_ * TK_];   // [r*TK_ + k]

    int tid  = threadIdx.x;
    int wv   = tid >> 5;                 // 0..4
    int lane = tid & 31;
    int ntN  = N / TN_;
    int mblk = blockIdx.x / ntN;
    int nblk = blockIdx.x - mblk * ntN;
    int m0   = mblk * TM_;
    int n0   = nblk * TN_;
    int ntilesN = N >> 4;
    int nt0  = n0 >> 4;                  // first 16-wide n-tile of this block

    v8f zero = {};
    v8f acc[4];
#pragma unroll
    for (int i = 0; i < 4; ++i) acc[i] = zero;

    int ar = tid >> 1;                   // 0..79 : A row within tile
    int ah = (tid & 1) << 4;             // 0 / 16 : A k-offset
    int khalf = (lane >> 4) << 3;        // 0 or 8
    const bf16_t* arow_lds = Asb + (((wv << 4) + (lane & 15)) * TK_);

    for (int k0 = 0; k0 < K; k0 += TK_) {
        int kt = k0 >> 5;
        if (k0 + TK_ < K) {              // overlap next K panel (global_prefetch_b8)
            __builtin_prefetch(A + (size_t)(m0 + ar) * lda + k0 + TK_ + ah, 0, 0);
            __builtin_prefetch(Bf + (((size_t)(kt + 1) * ntilesN + nt0) * 32 + lane) * 16, 0, 0);
        }
        // stage A panel (80x32): float4 quad load -> cvt -> two v8bf LDS stores
        const float4* Ap4 = (const float4*)(A + (size_t)(m0 + ar) * lda + k0 + ah);
        float4 f0 = Ap4[0], f1 = Ap4[1], f2 = Ap4[2], f3 = Ap4[3];
        *(v8bf*)(Asb + ar * TK_ + ah)     = cvt8(f0, f1);
        *(v8bf*)(Asb + ar * TK_ + ah + 8) = cvt8(f2, f3);
        __syncthreads();

        // A fragment: two aligned 16B LDS loads, combined in registers
        v8bf alo = *(const v8bf*)(arow_lds + khalf);
        v8bf ahi = *(const v8bf*)(arow_lds + khalf + 16);
        v16bf a = __builtin_shufflevector(alo, ahi,
                    0, 1, 2, 3, 4, 5, 6, 7, 8, 9, 10, 11, 12, 13, 14, 15);

        // B fragments straight from global (pre-swizzled, 32B contiguous per lane)
        const bf16_t* bbase = Bf + (((size_t)kt * ntilesN + nt0) * 32 + lane) * 16;
#pragma unroll
        for (int nt = 0; nt < 4; ++nt) {
            v16bf b = *(const v16bf*)(bbase + (size_t)nt * 32 * 16);
            acc[nt] = __builtin_amdgcn_wmma_f32_16x16x32_bf16(false, a, false, b,
                                                              (short)0, acc[nt],
                                                              false, false);
        }
        __syncthreads();
    }

    int rbase = m0 + (wv << 4) + ((lane >> 4) << 3);
    int ncl   = lane & 15;
#pragma unroll
    for (int nt = 0; nt < 4; ++nt) {
        int ncol = n0 + (nt << 4) + ncl;
        float bv = bias[ncol];
#pragma unroll
        for (int v = 0; v < 8; ++v)
            C[(size_t)(rbase + v) * N + ncol] = acc[nt][v] + bv;
    }
}

// ---------------- encoder GEMM: cur(BN,HID) = x[:,t] (BN,FIN) @ encW + enc_b ----
// K = 32 -> one WMMA per tile; A via float4 loads, B via fragment-layout global.
__global__ void encoder_gemm(const float* __restrict__ x,
                             const bf16_t* __restrict__ Wf,
                             const float* __restrict__ bias,
                             float* __restrict__ C, int t) {
    const int M = BN_, N = HID_;
    int wave = (blockIdx.x * blockDim.x + threadIdx.x) >> 5;
    int lane = threadIdx.x & 31;
    int ntN  = N >> 4;
    int mt   = wave / ntN;
    int nt   = wave - mt * ntN;
    if (mt >= (M >> 4)) return;                  // wave-uniform
    int m0 = mt << 4, n0 = nt << 4;
    int mrow  = m0 + (lane & 15);
    int khalf = (lane >> 4) << 3;

    int bb = mrow / N_;
    int nn = mrow - bb * N_;
    const float* Arow = x + ((size_t)(bb * T_ + t) * N_ + nn) * FIN_;

    const float4* Ap4 = (const float4*)(Arow + khalf);
    float4 f0 = Ap4[0], f1 = Ap4[1];
    const float4* Ap4h = (const float4*)(Arow + khalf + 16);
    float4 f2 = Ap4h[0], f3 = Ap4h[1];
    v16bf a = __builtin_shufflevector(cvt8(f0, f1), cvt8(f2, f3),
                0, 1, 2, 3, 4, 5, 6, 7, 8, 9, 10, 11, 12, 13, 14, 15);
    v16bf b = *(const v16bf*)(Wf + ((size_t)nt * 32 + lane) * 16);

    v8f acc = {};
    acc = __builtin_amdgcn_wmma_f32_16x16x32_bf16(false, a, false, b,
                                                  (short)0, acc, false, false);
    int ncol = n0 + (lane & 15);
    float bv = bias[ncol];
    int rbase = m0 + ((lane >> 4) << 3);
#pragma unroll
    for (int v = 0; v < 8; ++v)
        C[(size_t)(rbase + v) * N + ncol] = acc[v] + bv;
}

// ---------------- edge phase ----------------
__device__ __forceinline__ void atomicMaxF32(float* addr, float val) {
    if (val >= 0.f) atomicMax((int*)addr, __float_as_int(val));
    else            atomicMin((unsigned int*)addr, (unsigned int)__float_as_int(val));
}

__global__ void init_edge(float* __restrict__ m, float* __restrict__ s,
                          float* __restrict__ agg) {
    int i = blockIdx.x * blockDim.x + threadIdx.x;
    if (i < BN_ * H_) { m[i] = __int_as_float(0xFF800000); s[i] = 0.f; }
    if (i < BN_ * GHID_) agg[i] = 0.f;
}

// one wave per (edge,head) unit: logit = dot(leakyrelu(xl[src]+xr[dst]), att[head])
__global__ void edge_logit_max(const float* __restrict__ xl, const float* __restrict__ xr,
                               const long long* __restrict__ eidx,
                               const float* __restrict__ att,
                               float* __restrict__ logit, float* __restrict__ m) {
    int u    = (blockIdx.x * blockDim.x + threadIdx.x) >> 5;
    int lane = threadIdx.x & 31;
    if (u >= UNITS_) return;                      // wave-uniform
    int head = u & (H_ - 1);
    int eb   = u >> 1;
    int b    = eb / E_;
    int e    = eb - b * E_;
    long long src = eidx[e]      + (long long)b * N_;
    long long dst = eidx[E_ + e] + (long long)b * N_;

    const float4 vl = ((const float4*)(xl + (size_t)src * GHID_ + head * HID_))[lane];
    const float4 vr = ((const float4*)(xr + (size_t)dst * GHID_ + head * HID_))[lane];
    const float4 va = ((const float4*)(att + head * HID_))[lane];

    float s0 = vl.x + vr.x; s0 = s0 > 0.f ? s0 : NEG_SLOPE * s0;
    float s1 = vl.y + vr.y; s1 = s1 > 0.f ? s1 : NEG_SLOPE * s1;
    float s2 = vl.z + vr.z; s2 = s2 > 0.f ? s2 : NEG_SLOPE * s2;
    float s3 = vl.w + vr.w; s3 = s3 > 0.f ? s3 : NEG_SLOPE * s3;
    float sum = s0 * va.x + s1 * va.y + s2 * va.z + s3 * va.w;
#pragma unroll
    for (int o = 16; o > 0; o >>= 1) sum += __shfl_xor(sum, o, 32);
    if (lane == 0) {
        logit[u] = sum;
        atomicMaxF32(&m[(size_t)dst * H_ + head], sum);
    }
}

// one thread per unit: p = exp(logit - m[dst]); s[dst] += p
__global__ void edge_softmax(const float* __restrict__ logit,
                             const long long* __restrict__ eidx,
                             const float* __restrict__ m,
                             float* __restrict__ p, float* __restrict__ s) {
    int u = blockIdx.x * blockDim.x + threadIdx.x;
    if (u >= UNITS_) return;
    int head = u & (H_ - 1);
    int eb   = u >> 1;
    int b    = eb / E_;
    int e    = eb - b * E_;
    long long dst = eidx[E_ + e] + (long long)b * N_;
    float pv = __expf(logit[u] - m[(size_t)dst * H_ + head]);
    p[u] = pv;
    atomicAdd(&s[(size_t)dst * H_ + head], pv);
}

// one wave per unit: agg[dst] += (p/s[dst]) * xl[src]
__global__ void edge_agg(const float* __restrict__ xl,
                         const long long* __restrict__ eidx,
                         const float* __restrict__ p, const float* __restrict__ s,
                         float* __restrict__ agg) {
    int u    = (blockIdx.x * blockDim.x + threadIdx.x) >> 5;
    int lane = threadIdx.x & 31;
    if (u >= UNITS_) return;
    int head = u & (H_ - 1);
    int eb   = u >> 1;
    int b    = eb / E_;
    int e    = eb - b * E_;
    long long src = eidx[e]      + (long long)b * N_;
    long long dst = eidx[E_ + e] + (long long)b * N_;
    float alpha = p[u] / (s[(size_t)dst * H_ + head] + 1e-16f);
    const float4 v = ((const float4*)(xl + (size_t)src * GHID_ + head * HID_))[lane];
    float* base = agg + (size_t)dst * GHID_ + head * HID_ + lane * 4;
    atomicAdd(base + 0, alpha * v.x);
    atomicAdd(base + 1, alpha * v.y);
    atomicAdd(base + 2, alpha * v.z);
    atomicAdd(base + 3, alpha * v.w);
}

// spat = mean over heads + gat_bias
__global__ void spat_kernel(const float* __restrict__ agg, const float* __restrict__ bias,
                            float* __restrict__ spat) {
    int i = blockIdx.x * blockDim.x + threadIdx.x;
    if (i >= BN_ * HID_) return;
    int row = i >> 7, c = i & (HID_ - 1);
    spat[i] = 0.5f * (agg[(size_t)row * GHID_ + c] + agg[(size_t)row * GHID_ + HID_ + c])
              + bias[c];
}

// ---------------- fused GRU + LayerNorm ----------------
__global__ void gru_ln_kernel(const float* __restrict__ gx, const float* __restrict__ gh,
                              float* __restrict__ state, float* __restrict__ cur,
                              float* __restrict__ out,
                              const float* __restrict__ lg, const float* __restrict__ lb,
                              int l, int t) {
    int row = blockIdx.x;            // BN_ blocks
    int c   = threadIdx.x;           // HID_ threads
    size_t g = (size_t)row * G3_;
    float xr = gx[g + c], xz = gx[g + HID_ + c], xn = gx[g + 2 * HID_ + c];
    float hr = gh[g + c], hz = gh[g + HID_ + c], hn = gh[g + 2 * HID_ + c];
    float hprev = state[(size_t)row * (L_ * HID_) + l * HID_ + c];

    float r = 1.f / (1.f + __expf(-(xr + hr)));
    float z = 1.f / (1.f + __expf(-(xz + hz)));
    float n = tanhf(xn + r * hn);
    float h = (1.f - z) * n + z * hprev;

    __shared__ float s1[HID_], s2[HID_];
    s1[c] = h; s2[c] = h * h;
    __syncthreads();
#pragma unroll
    for (int o = HID_ / 2; o > 0; o >>= 1) {
        if (c < o) { s1[c] += s1[c + o]; s2[c] += s2[c + o]; }
        __syncthreads();
    }
    float mu  = s1[0] * (1.f / HID_);
    float var = s2[0] * (1.f / HID_) - mu * mu;
    float hn2 = (h - mu) * rsqrtf(var + LN_EPS) * lg[c] + lb[c];

    state[(size_t)row * (L_ * HID_) + l * HID_ + c] = hn2;
    cur[(size_t)row * HID_ + c] = hn2;
    int bb = row / N_, nn = row - bb * N_;
    out[((size_t)(bb * T_ + t) * N_ + nn) * (L_ * HID_) + l * HID_ + c] = hn2;
}

// ---------------- host orchestration ----------------
static inline size_t alignUp(size_t x) { return (x + 255) & ~(size_t)255; }

extern "C" void kernel_launch(void* const* d_in, const int* in_sizes, int n_in,
                              void* d_out, int out_size, void* d_ws, size_t ws_size,
                              hipStream_t stream) {
    const float*     x        = (const float*)d_in[0];
    const float*     enc_W    = (const float*)d_in[1];
    const float*     enc_b    = (const float*)d_in[2];
    const float*     gat_Wl   = (const float*)d_in[3];
    const float*     gat_Wr   = (const float*)d_in[4];
    const float*     gat_bl   = (const float*)d_in[5];
    const float*     gat_br   = (const float*)d_in[6];
    const float*     gat_att  = (const float*)d_in[7];
    const float*     gat_bias = (const float*)d_in[8];
    const float*     gru_Wih  = (const float*)d_in[9];
    const float*     gru_Whh  = (const float*)d_in[10];
    const float*     gru_bih  = (const float*)d_in[11];
    const float*     gru_bhh  = (const float*)d_in[12];
    const float*     ln_g     = (const float*)d_in[13];
    const float*     ln_b     = (const float*)d_in[14];
    const long long* eidx     = (const long long*)d_in[15];
    float* out = (float*)d_out;

    // ---- workspace layout ----
    char* w = (char*)d_ws;
    size_t off = 0;
    auto take = [&](size_t bytes) { size_t p = off; off = alignUp(off + bytes); return p; };

    bf16_t* encWb = (bf16_t*)(w + take(FIN_ * HID_ * sizeof(bf16_t)));
    bf16_t* WlB   = (bf16_t*)(w + take((size_t)L_ * HID_ * GHID_ * sizeof(bf16_t)));
    bf16_t* WrB   = (bf16_t*)(w + take((size_t)L_ * HID_ * GHID_ * sizeof(bf16_t)));
    bf16_t* WihT  = (bf16_t*)(w + take((size_t)L_ * HID_ * G3_ * sizeof(bf16_t)));
    bf16_t* WhhT  = (bf16_t*)(w + take((size_t)L_ * HID_ * G3_ * sizeof(bf16_t)));

    float* state = (float*)(w + take((size_t)BN_ * L_ * HID_ * sizeof(float)));
    float* cur   = (float*)(w + take((size_t)BN_ * HID_ * sizeof(float)));
    float* spat  = (float*)(w + take((size_t)BN_ * HID_ * sizeof(float)));

    // BIG region: xl|xr|agg during edge phase; gx|gh during GRU GEMMs (aliased)
    size_t bigOff = take((size_t)BN_ * 3 * GHID_ * sizeof(float)
                         + (size_t)(2 * UNITS_ + 2 * BN_ * H_) * sizeof(float));
    float* xl   = (float*)(w + bigOff);
    float* xr   = xl + (size_t)BN_ * GHID_;
    float* agg  = xr + (size_t)BN_ * GHID_;
    float* gx   = xl;                               // alias (xl/xr dead by then)
    float* gh   = xl + (size_t)BN_ * G3_;           // ends before logit region
    float* logit = xl + (size_t)BN_ * 3 * GHID_;
    float* pbuf  = logit + UNITS_;
    float* mbuf  = pbuf + UNITS_;
    float* sbuf  = mbuf + (size_t)BN_ * H_;

    // ---- per-call setup: swizzle all weights into WMMA fragment layout ----
    conv_bf16_frag<<<(FIN_ * HID_ + 255) / 256, 256, 0, stream>>>(
        enc_W, encWb, FIN_, HID_, 0);
    for (int l = 0; l < L_; ++l) {
        conv_bf16_frag<<<((HID_ * GHID_) + 255) / 256, 256, 0, stream>>>(
            gat_Wl + (size_t)l * HID_ * GHID_, WlB + (size_t)l * HID_ * GHID_,
            HID_, GHID_, 0);
        conv_bf16_frag<<<((HID_ * GHID_) + 255) / 256, 256, 0, stream>>>(
            gat_Wr + (size_t)l * HID_ * GHID_, WrB + (size_t)l * HID_ * GHID_,
            HID_, GHID_, 0);
        conv_bf16_frag<<<((HID_ * G3_) + 255) / 256, 256, 0, stream>>>(
            gru_Wih + (size_t)l * G3_ * HID_, WihT + (size_t)l * HID_ * G3_,
            HID_, G3_, 1);
        conv_bf16_frag<<<((HID_ * G3_) + 255) / 256, 256, 0, stream>>>(
            gru_Whh + (size_t)l * G3_ * HID_, WhhT + (size_t)l * HID_ * G3_,
            HID_, G3_, 1);
    }
    hipMemsetAsync(state, 0, (size_t)BN_ * L_ * HID_ * sizeof(float), stream);

    const int encBlocks  = ((BN_ / 16) * (HID_ / 16) * 32 + 255) / 256;   // 1250
    const int gatBlocks  = (BN_ / TM_) * (GHID_ / TN_);                   // 250*4
    const int gruBlocks  = (BN_ / TM_) * (G3_ / TN_);                     // 250*6
    const int initBlocks = (BN_ * GHID_ + 255) / 256;
    const int waveUBlks  = (UNITS_ * 32 + 255) / 256;                     // 40000
    const int thrUBlks   = (UNITS_ + 255) / 256;
    const int spatBlocks = (BN_ * HID_ + 255) / 256;

    for (int t = 0; t < T_; ++t) {
        encoder_gemm<<<encBlocks, 256, 0, stream>>>(x, encWb, enc_b, cur, t);

        for (int l = 0; l < L_; ++l) {
            // xl / xr projections
            wmma_gemm_bias<<<gatBlocks, GEMM_THREADS, 0, stream>>>(
                cur, HID_, WlB + (size_t)l * HID_ * GHID_, gat_bl + l * GHID_,
                xl, BN_, GHID_, HID_);
            wmma_gemm_bias<<<gatBlocks, GEMM_THREADS, 0, stream>>>(
                cur, HID_, WrB + (size_t)l * HID_ * GHID_, gat_br + l * GHID_,
                xr, BN_, GHID_, HID_);

            // edge attention (segment max / softmax / weighted sum)
            init_edge<<<initBlocks, 256, 0, stream>>>(mbuf, sbuf, agg);
            edge_logit_max<<<waveUBlks, 256, 0, stream>>>(
                xl, xr, eidx, gat_att + (size_t)l * H_ * HID_, logit, mbuf);
            edge_softmax<<<thrUBlks, 256, 0, stream>>>(logit, eidx, mbuf, pbuf, sbuf);
            edge_agg<<<waveUBlks, 256, 0, stream>>>(xl, eidx, pbuf, sbuf, agg);
            spat_kernel<<<spatBlocks, 256, 0, stream>>>(agg, gat_bias + l * HID_, spat);

            // GRU gate GEMMs (gx/gh alias the xl/xr region — now dead)
            wmma_gemm_bias<<<gruBlocks, GEMM_THREADS, 0, stream>>>(
                spat, HID_, WihT + (size_t)l * HID_ * G3_, gru_bih + l * G3_,
                gx, BN_, G3_, HID_);
            wmma_gemm_bias<<<gruBlocks, GEMM_THREADS, 0, stream>>>(
                state + l * HID_, L_ * HID_, WhhT + (size_t)l * HID_ * G3_, gru_bhh + l * G3_,
                gh, BN_, G3_, HID_);

            gru_ln_kernel<<<BN_, HID_, 0, stream>>>(
                gx, gh, state, cur, out, ln_g + l * HID_, ln_b + l * HID_, l, t);
        }
    }
    (void)in_sizes; (void)n_in; (void)out_size; (void)ws_size;
}
